// R_RNN_Layer_6382321402090
// MI455X (gfx1250) — matrix operand order repaired and checked by reading, not compiled, not run on hardware
//
#include <hip/hip_runtime.h>
#include <hip/hip_bf16.h>

// ---------------------------------------------------------------------------
// Problem constants (from the reference): B=8, T=4096, F=512, C=128, U=256, O=64
// ---------------------------------------------------------------------------
#define BB 8
#define TT 4096
#define FF 512
#define CC 128
#define UU 256
#define OO 64
#define G4U 1024          // 4*U
#define MROWS (BB * TT)   // 32768
#define WSTR 264          // LDS row stride in bf16 (256 + 8 pad = 528 B)

typedef __bf16 bf16_t;
typedef bf16_t v16bf __attribute__((ext_vector_type(16)));
typedef bf16_t v8bf  __attribute__((ext_vector_type(8)));
typedef float  v8f   __attribute__((ext_vector_type(8)));

// ---------------------------------------------------------------------------
// Helpers
// ---------------------------------------------------------------------------
__device__ __forceinline__ v16bf ld_frag16(const bf16_t* p0, const bf16_t* p1) {
    v8bf lo = *(const v8bf*)p0;
    v8bf hi = *(const v8bf*)p1;
    v16bf r;
#pragma unroll
    for (int i = 0; i < 8; ++i) { r[i] = lo[i]; r[8 + i] = hi[i]; }
    return r;
}

__device__ __forceinline__ float fast_sigmoid(float x) {
    return 1.0f / (1.0f + __expf(-x));
}
__device__ __forceinline__ float fast_tanh(float x) {
    // tanh(x) = 1 - 2/(exp(2x)+1); saturates correctly at +-1
    return 1.0f - 2.0f / (__expf(2.0f * x) + 1.0f);
}

// ---------------------------------------------------------------------------
// Transpose + fp32 -> bf16 convert:   dst[c][r] = (bf16) src[r][c]
// ---------------------------------------------------------------------------
__global__ void transpose_to_bf16(const float* __restrict__ src,
                                  bf16_t* __restrict__ dst, int R, int Cn) {
    int idx = blockIdx.x * blockDim.x + threadIdx.x;
    if (idx >= R * Cn) return;
    int r = idx / Cn;
    int c = idx - r * Cn;
    dst[(size_t)c * R + r] = (bf16_t)src[idx];
}

// ---------------------------------------------------------------------------
// Generic WMMA GEMM:  D[M,N] = A[M,K] * Bt[N,K]^T + bias[N]
// One wave per 16x16 output tile, K in steps of 32 (bf16 WMMA).
//   AMODE: 0 = A bf16 row-major [M,K]
//          1 = A fp32 row-major [M,K] (converted on the fly)
//          2 = A bf16 in scan layout [t][K][B], row = b*T + t
//   DMODE: 0 = D fp32 row-major [M,N]
//          1 = D bf16 row-major [M,N]
//          2 = D bf16 in scan layout [t][N][B], row = b*T + t
// ---------------------------------------------------------------------------
template <int AMODE, int DMODE, int K>
__global__ __launch_bounds__(256) void gemm_wmma(const void* __restrict__ Av,
                                                 const bf16_t* __restrict__ Bt,
                                                 const float* __restrict__ bias,
                                                 void* __restrict__ Dv,
                                                 int M, int N) {
    const int wv   = blockIdx.x * (blockDim.x >> 5) + (threadIdx.x >> 5);
    const int lane = threadIdx.x & 31;
    const int half = lane >> 4;
    const int l15  = lane & 15;

    const int ntiles = N >> 4;
    const int mt = wv / ntiles;
    const int nt = wv - mt * ntiles;
    if (mt * 16 >= M) return;  // uniform per wave

    const int arow = mt * 16 + l15;       // A row this lane reads
    const int ncol = nt * 16 + l15;       // B/D column this lane owns

    v8f acc = {};

#pragma unroll 4
    for (int ks = 0; ks < K; ks += 32) {
        // ---- A fragment: 16-bit A 16x32 layout ----
        v16bf a;
        const int ka = ks + half * 8;
        if constexpr (AMODE == 1) {
            const float* pa = (const float*)Av + (size_t)arow * K + ka;
#pragma unroll
            for (int i = 0; i < 8; ++i) {
                a[i]     = (bf16_t)pa[i];
                a[8 + i] = (bf16_t)pa[16 + i];
            }
        } else if constexpr (AMODE == 2) {
            const bf16_t* H = (const bf16_t*)Av;
            const int b  = arow / TT;
            const int th = arow & (TT - 1);
#pragma unroll
            for (int i = 0; i < 8; ++i) {
                a[i]     = H[((size_t)th * UU + ka + i) * BB + b];
                a[8 + i] = H[((size_t)th * UU + ka + 16 + i) * BB + b];
            }
        } else {
            const bf16_t* pa = (const bf16_t*)Av + (size_t)arow * K + ka;
            a = ld_frag16(pa, pa + 16);
        }
        // ---- B fragment: K values {kb..kb+7, kb+16..kb+23} of column `ncol` ----
        const int kb = ks + half * 8;
        const bf16_t* pb = Bt + (size_t)ncol * K + kb;
        v16bf b = ld_frag16(pb, pb + 16);

        acc = __builtin_amdgcn_wmma_f32_16x16x32_bf16(
            false, a, false, b, (short)0, acc, false, false);
    }

    const float bv = bias ? bias[ncol] : 0.0f;
#pragma unroll
    for (int v = 0; v < 8; ++v) {
        const int row = mt * 16 + v + half * 8;
        const float val = acc[v] + bv;
        if constexpr (DMODE == 0) {
            ((float*)Dv)[(size_t)row * N + ncol] = val;
        } else if constexpr (DMODE == 1) {
            ((bf16_t*)Dv)[(size_t)row * N + ncol] = (bf16_t)val;
        } else {
            const int b  = row / TT;
            const int th = row & (TT - 1);
            ((bf16_t*)Dv)[((size_t)th * N + ncol) * BB + b] = (bf16_t)val;
        }
    }
}

// ---------------------------------------------------------------------------
// Sequential LSTM scan. One workgroup = 16 waves (512 threads).
// Wave w owns cell-state columns [16w, 16w+16). Per step it computes all four
// gate tiles with 8 bf16 WMMAs each (K=256).
//
// Residency / layout decisions for the serial critical path:
//   gates i,f   -> preloaded into 128 VGPRs per wave (wi[8], wf[8])
//   gates g,o   -> staged once into LDS (pad stride 264 -> conflict-free b128)
//   h           -> double-buffered LDS tile, rows padded to 264 as well
//   z           -> [t][4U][B] layout: 4 coalesced b128 loads/lane/step,
//                  issued before the WMMA loop to hide global latency
//   hs          -> [t][U][B] layout: 1 b128 store/lane/step
// Total LDS: ~287 KB of the WGP's 320 KB.
// ---------------------------------------------------------------------------
__global__ __launch_bounds__(512, 1) void lstm_scan(const bf16_t* __restrict__ z,
                                                    const bf16_t* __restrict__ rt,
                                                    bf16_t* __restrict__ hs) {
    __shared__ __align__(16) bf16_t hbuf[2][16][WSTR];
    __shared__ __align__(16) bf16_t wlds[2 * UU][WSTR];  // gates g,o: rows n-512

    const int tid  = threadIdx.x;
    const int wave = tid >> 5;        // 0..15
    const int lane = tid & 31;
    const int half = lane >> 4;
    const int l15  = lane & 15;
    const int wcol = wave * 16;       // this wave's 16-column slice of U
    const bool act = (lane < 16);     // lanes holding valid batch rows 0..7

    // Stage gates g,o (global rows 512..1023 of rt) into LDS, b128 chunks.
    for (int idx = tid; idx < 2 * UU * (UU / 8); idx += 512) {
        const int row = idx >> 5;     // 32 chunks of 8 bf16 per 256-wide row
        const int ch  = idx & 31;
        *(v8bf*)&wlds[row][ch * 8] =
            *(const v8bf*)&rt[(size_t)(2 * UU + row) * UU + ch * 8];
    }
    // Zero both h buffers (h0 = 0; rows 8..15 stay zero forever).
    for (int i = tid; i < 2 * 16 * WSTR; i += 512)
        ((bf16_t*)hbuf)[i] = (bf16_t)0.0f;

    // Preload gates i,f weight fragments into registers (128 VGPRs).
    v16bf wi[8], wf[8];
#pragma unroll
    for (int ks = 0; ks < 8; ++ks) {
        const int kb = ks * 32 + half * 8;
        const bf16_t* p0 = rt + (size_t)(0 * UU + wcol + l15) * UU + kb;
        const bf16_t* p1 = rt + (size_t)(1 * UU + wcol + l15) * UU + kb;
        wi[ks] = ld_frag16(p0, p0 + 16);
        wf[ks] = ld_frag16(p1, p1 + 16);
    }
    __syncthreads();

    v8f cst = {};  // cell state tile (rows 8..15 identically zero)

    for (int t = 0; t < TT; ++t) {
        const int p = t & 1;

        // Issue input-contribution loads early; consumed after the WMMA loop.
        v8bf zv0 = {}, zv1 = {}, zv2 = {}, zv3 = {};
        if (act) {
            const bf16_t* zp = z + ((size_t)t * G4U + wcol + l15) * BB;
            zv0 = *(const v8bf*)(zp + (size_t)(0 * UU) * BB);
            zv1 = *(const v8bf*)(zp + (size_t)(1 * UU) * BB);
            zv2 = *(const v8bf*)(zp + (size_t)(2 * UU) * BB);
            zv3 = *(const v8bf*)(zp + (size_t)(3 * UU) * BB);
        }

        v8f a0 = {}, a1 = {}, a2 = {}, a3 = {};

#pragma unroll
        for (int ks = 0; ks < 8; ++ks) {
            // A fragment from h in LDS (16-bit A 16x32 layout)
            const bf16_t* pa = &hbuf[p][l15][ks * 32 + half * 8];
            v16bf a = ld_frag16(pa, pa + 16);

            const int kb = ks * 32 + half * 8;
            const bf16_t* pg = &wlds[0 * UU + wcol + l15][kb];
            const bf16_t* po = &wlds[1 * UU + wcol + l15][kb];

            a0 = __builtin_amdgcn_wmma_f32_16x16x32_bf16(
                false, a, false, wi[ks], (short)0, a0, false, false);
            a1 = __builtin_amdgcn_wmma_f32_16x16x32_bf16(
                false, a, false, wf[ks], (short)0, a1, false, false);
            a2 = __builtin_amdgcn_wmma_f32_16x16x32_bf16(
                false, a, false, ld_frag16(pg, pg + 16), (short)0, a2, false, false);
            a3 = __builtin_amdgcn_wmma_f32_16x16x32_bf16(
                false, a, false, ld_frag16(po, po + 16), (short)0, a3, false, false);
        }

        // Add the precomputed input contribution (valid batch rows only).
        if (act) {
#pragma unroll
            for (int v = 0; v < 8; ++v) {
                a0[v] += (float)zv0[v];
                a1[v] += (float)zv1[v];
                a2[v] += (float)zv2[v];
                a3[v] += (float)zv3[v];
            }
        }

        // Gates + state update + write next h.
        bf16_t hvv[8];
#pragma unroll
        for (int v = 0; v < 8; ++v) {
            const float ig = fast_sigmoid(a0[v]);
            const float fg = fast_sigmoid(a1[v]);
            const float gg = fast_tanh(a2[v]);
            const float og = fast_sigmoid(a3[v]);
            const float cv = fg * cst[v] + ig * gg;
            cst[v] = cv;
            const float hv = og * fast_tanh(cv);
            hvv[v] = (bf16_t)hv;

            const int m = v + half * 8;
            hbuf[p ^ 1][m][wcol + l15] = hvv[v];
        }
        // One coalesced 16B store of this lane's 8 batch values.
        if (act) {
            v8bf hv8;
#pragma unroll
            for (int v = 0; v < 8; ++v) hv8[v] = hvv[v];
            *(v8bf*)&hs[((size_t)t * UU + wcol + l15) * BB] = hv8;
        }
        __syncthreads();
    }
}

// ---------------------------------------------------------------------------
// Host-side orchestration
// ---------------------------------------------------------------------------
extern "C" void kernel_launch(void* const* d_in, const int* in_sizes, int n_in,
                              void* d_out, int out_size, void* d_ws, size_t ws_size,
                              hipStream_t stream) {
    (void)in_sizes; (void)n_in; (void)out_size; (void)ws_size;

    const float* x      = (const float*)d_in[0];  // [B,T,F]
    const float* w_in   = (const float*)d_in[1];  // [F,C]
    const float* b_in   = (const float*)d_in[2];  // [C]
    const float* kernel = (const float*)d_in[3];  // [C,4U]
    const float* rec_k  = (const float*)d_in[4];  // [U,4U]
    const float* bias   = (const float*)d_in[5];  // [4U]
    const float* w_out  = (const float*)d_in[6];  // [U,O]
    const float* b_out  = (const float*)d_in[7];  // [O]
    float* out = (float*)d_out;                   // [B,T,O]

    // Workspace layout (256-byte aligned slices)
    char* ws = (char*)d_ws;
    size_t off = 0;
    auto carve = [&](size_t bytes) {
        char* p = ws + off;
        off = (off + bytes + 255) & ~(size_t)255;
        return p;
    };
    bf16_t* w_in_t   = (bf16_t*)carve((size_t)CC * FF * 2);     // [C,F]
    bf16_t* kernel_t = (bf16_t*)carve((size_t)G4U * CC * 2);    // [4U,C]
    bf16_t* rec_t    = (bf16_t*)carve((size_t)G4U * UU * 2);    // [4U,U]
    bf16_t* w_out_t  = (bf16_t*)carve((size_t)OO * UU * 2);     // [O,U]
    bf16_t* xr       = (bf16_t*)carve((size_t)MROWS * CC * 2);  // [B*T,C]
    bf16_t* zbuf     = (bf16_t*)carve((size_t)MROWS * G4U * 2); // [t][4U][B]
    bf16_t* hsbuf    = (bf16_t*)carve((size_t)MROWS * UU * 2);  // [t][U][B]

    // 1) Transpose weights to [N][K] bf16 for the WMMA B-operand layout.
    transpose_to_bf16<<<(FF * CC + 255) / 256, 256, 0, stream>>>(w_in, w_in_t, FF, CC);
    transpose_to_bf16<<<(CC * G4U + 255) / 256, 256, 0, stream>>>(kernel, kernel_t, CC, G4U);
    transpose_to_bf16<<<(UU * G4U + 255) / 256, 256, 0, stream>>>(rec_k, rec_t, UU, G4U);
    transpose_to_bf16<<<(UU * OO + 255) / 256, 256, 0, stream>>>(w_out, w_out_t, UU, OO);

    // 2) xr = x @ w_in + b_in          [32768,512]x[512,128] -> bf16
    {
        const int waves = (MROWS / 16) * (CC / 16);
        gemm_wmma<1, 1, FF><<<waves / 8, 256, 0, stream>>>(
            x, w_in_t, b_in, xr, MROWS, CC);
    }
    // 3) z = xr @ kernel + bias        [32768,128]x[128,1024] -> bf16 [t][4U][B]
    {
        const int waves = (MROWS / 16) * (G4U / 16);
        gemm_wmma<0, 2, CC><<<waves / 8, 256, 0, stream>>>(
            xr, kernel_t, bias, zbuf, MROWS, G4U);
    }
    // 4) Sequential LSTM recurrence    -> hs bf16 [t][U][B]
    lstm_scan<<<1, 512, 0, stream>>>(zbuf, rec_t, hsbuf);

    // 5) out = hs @ w_out + b_out      [32768,256]x[256,64] -> fp32
    {
        const int waves = (MROWS / 16) * (OO / 16);
        gemm_wmma<2, 0, UU><<<waves / 8, 256, 0, stream>>>(
            hsbuf, w_out_t, b_out, out, MROWS, OO);
    }
}
